// LSTMFootballPredictionModel_71081708749388
// MI455X (gfx1250) — matrix operand order repaired and checked by reading, not compiled, not run on hardware
//
#include <hip/hip_runtime.h>
#include <hip/hip_bf16.h>
#include <stdint.h>

// ---------------------------------------------------------------------------
// Types
// ---------------------------------------------------------------------------
typedef __attribute__((ext_vector_type(16))) __bf16 v16bf;
typedef __attribute__((ext_vector_type(8)))  float  v8f;

constexpr int kE    = 9;
constexpr int kH    = 256;
constexpr int kG    = 1024;   // 4*H gate dims (i,f,g,o)
constexpr int kT    = 42;     // sequence length
constexpr int kNSeq = 2048;   // B*2 sequences
constexpr int kXK   = 32;     // E padded to one WMMA K-chunk
constexpr int kHPAD = 272;    // halves per LDS h row (padded vs 256 for banks)

union FragBF {
  v16bf v;
  uint4 q[2];
};

__device__ __forceinline__ unsigned short f2bf(float f) {
  unsigned int u = __float_as_uint(f);
  u += 0x7FFFu + ((u >> 16) & 1u);   // round-to-nearest-even
  return (unsigned short)(u >> 16);
}
__device__ __forceinline__ float bf2f(unsigned int h) {
  return __uint_as_float(h << 16);
}
__device__ __forceinline__ uint4 pack8bf(v8f a) {
  uint4 r;
  r.x = (unsigned)f2bf(a[0]) | ((unsigned)f2bf(a[1]) << 16);
  r.y = (unsigned)f2bf(a[2]) | ((unsigned)f2bf(a[3]) << 16);
  r.z = (unsigned)f2bf(a[4]) | ((unsigned)f2bf(a[5]) << 16);
  r.w = (unsigned)f2bf(a[6]) | ((unsigned)f2bf(a[7]) << 16);
  return r;
}
__device__ __forceinline__ float fsig(float x) { return 1.0f / (1.0f + __expf(-x)); }
__device__ __forceinline__ float ftanh(float x) {
  float e = __expf(2.0f * x);
  return (e - 1.0f) / (e + 1.0f);
}

// 32B global B-fragment load; `base` keeps addrspace(1) inference so these
// lower to global_load_b128 (saddr + voffset), tracked by LOADcnt only.
__device__ __forceinline__ FragBF loadB(const unsigned short* __restrict__ base, int off) {
  FragBF f;
  f.q[0] = *(const uint4*)(base + off);
  f.q[1] = *(const uint4*)(base + off + 8);
  return f;
}

// ---------------------------------------------------------------------------
// Prep kernels: bf16 conversions + sequence assembly (gather + interleave)
// ---------------------------------------------------------------------------
__global__ void k_prep_whh(const float* __restrict__ whh_f,
                           unsigned short* __restrict__ whh_bf) {
  int i = blockIdx.x * blockDim.x + threadIdx.x;   // 0 .. 1024*256-1
  whh_bf[i] = f2bf(whh_f[i]);
}

__global__ void k_prep_wih_bias(const float* __restrict__ wih_f,
                                const float* __restrict__ bih,
                                const float* __restrict__ bhh,
                                unsigned short* __restrict__ wih_bf,
                                float* __restrict__ biasv) {
  int g = blockIdx.x * blockDim.x + threadIdx.x;
  if (g < kG) {
    biasv[g] = bih[g] + bhh[g];
    #pragma unroll
    for (int k = 0; k < kXK; ++k)
      wih_bf[g * kXK + k] = (k < kE) ? f2bf(wih_f[g * kE + k]) : (unsigned short)0;
  }
}

// Build x sequence: interleave(stats[1..20], player_emb) ++ stats[0] ++ team_emb
// zero-padded to K=32 halves per (seq, t) so it drops straight into A-fragments.
__global__ void k_prep_seq(const float* __restrict__ stats,
                           const int*   __restrict__ team_ids,
                           const int*   __restrict__ player_ids,
                           const float* __restrict__ team_emb,
                           const float* __restrict__ player_emb,
                           unsigned short* __restrict__ xbf) {
  int idx = blockIdx.x * blockDim.x + threadIdx.x;  // over 2048*42 positions
  if (idx >= kNSeq * kT) return;
  int s = idx / kT;
  int t = idx - s * kT;
  const float* src;
  if (t < 40) {
    int p = t >> 1;
    if ((t & 1) == 0) src = stats + (size_t)(s * 21 + 1 + p) * kE;
    else              src = player_emb + (size_t)player_ids[s * 20 + p] * kE;
  } else if (t == 40) {
    src = stats + (size_t)(s * 21) * kE;
  } else {
    src = team_emb + (size_t)team_ids[s] * kE;
  }
  unsigned short* dst = xbf + (size_t)idx * kXK;
  #pragma unroll
  for (int k = 0; k < kE; ++k) dst[k] = f2bf(src[k]);
  #pragma unroll
  for (int k = kE; k < kXK; ++k) dst[k] = 0;
}

// ---------------------------------------------------------------------------
// Main fused LSTM kernel: 1 block = 16 sequences, 32 waves, full 42-step
// recurrence + output head. bf16 WMMA 16x16x32 for x-proj and h-proj GEMMs.
// W_hh B-fragments stream from the shared L2-resident bf16 copy each step
// (opaque VGPR offset defeats LICM without losing global addrspace), with a
// one-deep software pipeline so multiple load groups stay in flight.
// ---------------------------------------------------------------------------
__global__ __launch_bounds__(1024, 1) void lstm_wmma_kernel(
    const unsigned short* __restrict__ xbf,    // [2048][42][32] bf16
    const unsigned short* __restrict__ whh,    // [1024][256]    bf16
    const unsigned short* __restrict__ wih,    // [1024][32]     bf16 (padded)
    const float* __restrict__ biasv,           // [1024] = b_ih + b_hh
    const float* __restrict__ Wout,            // [3][512]
    const float* __restrict__ bout,            // [3]
    float* __restrict__ out)                   // [1024][3]
{
  __shared__ unsigned short h_lds[16 * kHPAD];   // bf16 hidden state, row-major
  __shared__ unsigned short gates[kG * 16];      // bf16 pre-activations [g][row]
  __shared__ float logit_l[8][3];

  const int tid  = threadIdx.x;
  const int lane = tid & 31;
  const int wv   = tid >> 5;          // wave 0..31, owns 32 gate columns
  const int mrow = lane & 15;
  const int kh   = lane >> 4;         // lane half-select (0/1)
  const int base_seq = blockIdx.x * 16;
  const int col0 = wv * 32;

  // Per-lane invariant offsets into W_hh for this wave's two column tiles.
  const int wrow0 = (col0 + mrow) * kH + kh * 16;        // halves
  const int wrow1 = (col0 + 16 + mrow) * kH + kh * 16;   // halves

  // Persistent W_ih B-fragments (one K=32 chunk covers the padded E=9 input).
  FragBF bx0 = loadB(wih, (col0 + mrow) * kXK + kh * 16);
  FragBF bx1 = loadB(wih, (col0 + 16 + mrow) * kXK + kh * 16);
  const float bias0 = biasv[col0 + mrow];
  const float bias1 = biasv[col0 + 16 + mrow];

  // Per-thread LSTM cell state: column kcol, rows r0..r0+3 of the 16-seq tile.
  const int kcol = tid >> 2;          // 0..255
  const int r0   = (tid & 3) * 4;     // 0,4,8,12
  float cstv[4] = {0.f, 0.f, 0.f, 0.f};

  for (int i = tid; i < 16 * kHPAD; i += 1024) h_lds[i] = 0;

  for (int t = 0; t < kT; ++t) {
    __syncthreads();   // h_lds ready (init or previous step's update)

    if (t + 1 < kT)    // speculative prefetch of next step's x tile
      __builtin_prefetch(xbf + (size_t)((base_seq + mrow) * kT + t + 1) * kXK, 0, 0);

    // x A-fragment (16 seqs x K=32, only K<9 nonzero)
    FragBF ax;
    {
      const unsigned short* xp = xbf + (size_t)((base_seq + mrow) * kT + t) * kXK + kh * 16;
      ax.q[0] = *(const uint4*)(xp);
      ax.q[1] = *(const uint4*)(xp + 8);
    }

    v8f acc0, acc1;
    #pragma unroll
    for (int j = 0; j < 8; ++j) { acc0[j] = bias0; acc1[j] = bias1; }

    // gates += x_t @ W_ih^T   (one padded K=32 WMMA per tile)
    acc0 = __builtin_amdgcn_wmma_f32_16x16x32_bf16(false, ax.v, false, bx0.v,
                                                   (short)0, acc0, false, false);
    acc1 = __builtin_amdgcn_wmma_f32_16x16x32_bf16(false, ax.v, false, bx1.v,
                                                   (short)0, acc1, false, false);

    // Optimization barrier on the per-lane VGPR offsets only: LICM cannot
    // hoist the weight loads (no 128-VGPR persistence, no scratch spills),
    // but `whh` itself keeps its global address space -> global_load_b128.
    int o0 = wrow0, o1 = wrow1;
    asm volatile("" : "+v"(o0), "+v"(o1));

    // gates += h_{t-1} @ W_hh^T  (K=256 in 8 chunks; W_hh streams from L2).
    // One-deep pipeline: chunk kf+1's B-fragments are issued before the
    // WMMAs of chunk kf so >=2 load groups are always in flight.
    FragBF b0 = loadB(whh, o0);
    FragBF b1 = loadB(whh, o1);
    #pragma unroll
    for (int kf = 0; kf < 8; ++kf) {
      FragBF n0, n1;
      if (kf < 7) {
        n0 = loadB(whh, o0 + (kf + 1) * 32);
        n1 = loadB(whh, o1 + (kf + 1) * 32);
      }
      FragBF ah;
      const unsigned short* hp = h_lds + mrow * kHPAD + kf * 32 + kh * 16;
      ah.q[0] = *(const uint4*)(hp);
      ah.q[1] = *(const uint4*)(hp + 8);
      acc0 = __builtin_amdgcn_wmma_f32_16x16x32_bf16(false, ah.v, false, b0.v,
                                                     (short)0, acc0, false, false);
      acc1 = __builtin_amdgcn_wmma_f32_16x16x32_bf16(false, ah.v, false, b1.v,
                                                     (short)0, acc1, false, false);
      if (kf < 7) { b0 = n0; b1 = n1; }
    }

    // Store pre-activations. C/D layout: VGPR j -> row j + kh*8, col mrow.
    *(uint4*)(gates + (col0 + mrow) * 16 + kh * 8)      = pack8bf(acc0);
    *(uint4*)(gates + (col0 + 16 + mrow) * 16 + kh * 8) = pack8bf(acc1);
    __syncthreads();

    // Elementwise cell update (gate order i,f,g,o), h written back as bf16.
    uint2 ui = *(const uint2*)(gates + (kcol      ) * 16 + r0);
    uint2 uf = *(const uint2*)(gates + (kcol + 256) * 16 + r0);
    uint2 ug = *(const uint2*)(gates + (kcol + 512) * 16 + r0);
    uint2 uo = *(const uint2*)(gates + (kcol + 768) * 16 + r0);
    float gi[4] = { bf2f(ui.x & 0xFFFFu), bf2f(ui.x >> 16), bf2f(ui.y & 0xFFFFu), bf2f(ui.y >> 16) };
    float gf[4] = { bf2f(uf.x & 0xFFFFu), bf2f(uf.x >> 16), bf2f(uf.y & 0xFFFFu), bf2f(uf.y >> 16) };
    float gg[4] = { bf2f(ug.x & 0xFFFFu), bf2f(ug.x >> 16), bf2f(ug.y & 0xFFFFu), bf2f(ug.y >> 16) };
    float go[4] = { bf2f(uo.x & 0xFFFFu), bf2f(uo.x >> 16), bf2f(uo.y & 0xFFFFu), bf2f(uo.y >> 16) };
    #pragma unroll
    for (int e = 0; e < 4; ++e) {
      float cv = fsig(gf[e]) * cstv[e] + fsig(gi[e]) * ftanh(gg[e]);
      cstv[e] = cv;
      float hv = fsig(go[e]) * ftanh(cv);
      h_lds[(r0 + e) * kHPAD + kcol] = f2bf(hv);
    }
  }

  // ---- fused head: logits = concat(h_home, h_away) @ W_out^T + b_out, softmax
  __syncthreads();
  if (tid < 24) {
    int m = tid / 3, j = tid - 3 * m;
    float s = bout[j];
    for (int k = 0; k < 512; ++k) {
      int r = 2 * m + (k >> 8);
      s += bf2f((unsigned int)h_lds[r * kHPAD + (k & 255)]) * Wout[j * 512 + k];
    }
    logit_l[m][j] = s;
  }
  __syncthreads();
  if (tid < 8) {
    float a = logit_l[tid][0], b = logit_l[tid][1], c = logit_l[tid][2];
    float mx = fmaxf(a, fmaxf(b, c));
    float ea = __expf(a - mx), eb = __expf(b - mx), ec = __expf(c - mx);
    float inv = 1.0f / (ea + eb + ec);
    float* op = out + (size_t)(blockIdx.x * 8 + tid) * 3;
    op[0] = ea * inv; op[1] = eb * inv; op[2] = ec * inv;
  }
}

// ---------------------------------------------------------------------------
// Launch
// ---------------------------------------------------------------------------
extern "C" void kernel_launch(void* const* d_in, const int* in_sizes, int n_in,
                              void* d_out, int out_size, void* d_ws, size_t ws_size,
                              hipStream_t stream) {
  (void)in_sizes; (void)n_in; (void)out_size; (void)ws_size;
  const float* stats      = (const float*)d_in[0];
  const int*   team_ids   = (const int*)  d_in[1];
  const int*   player_ids = (const int*)  d_in[2];
  const float* team_emb   = (const float*)d_in[3];
  const float* player_emb = (const float*)d_in[4];
  const float* W_ih       = (const float*)d_in[5];
  const float* W_hh       = (const float*)d_in[6];
  const float* b_ih       = (const float*)d_in[7];
  const float* b_hh       = (const float*)d_in[8];
  const float* W_out      = (const float*)d_in[9];
  const float* b_out      = (const float*)d_in[10];
  float* out = (float*)d_out;

  // Workspace layout (256B-aligned offsets):
  char* ws = (char*)d_ws;
  unsigned short* xbf   = (unsigned short*)(ws);                              // 2048*42*32*2 = 5,505,024 B
  unsigned short* whhbf = (unsigned short*)(ws + 5505024);                    // 1024*256*2  =   524,288 B
  unsigned short* wihbf = (unsigned short*)(ws + 5505024 + 524288);           // 1024*32*2   =    65,536 B
  float*          biasv = (float*)        (ws + 5505024 + 524288 + 65536);    // 1024*4      =     4,096 B

  hipLaunchKernelGGL(k_prep_whh, dim3((kG * kH) / 256), dim3(256), 0, stream, W_hh, whhbf);
  hipLaunchKernelGGL(k_prep_wih_bias, dim3(4), dim3(256), 0, stream, W_ih, b_ih, b_hh, wihbf, biasv);
  hipLaunchKernelGGL(k_prep_seq, dim3((kNSeq * kT + 255) / 256), dim3(256), 0, stream,
                     stats, team_ids, player_ids, team_emb, player_emb, xbf);
  hipLaunchKernelGGL(lstm_wmma_kernel, dim3(kNSeq / 16), dim3(1024), 0, stream,
                     xbf, whhbf, wihbf, biasv, W_out, b_out, out);
}
